// CogVideoXAttention_13640816132846
// MI455X (gfx1250) — matrix operand order repaired and checked by reading, not compiled, not run on hardware
//
#include <hip/hip_runtime.h>
#include <hip/hip_fp16.h>

typedef __attribute__((ext_vector_type(16))) _Float16 v16h;
typedef __attribute__((ext_vector_type(8)))  float    v8f;
typedef __attribute__((ext_vector_type(4)))  int      v4i;

#define D_MODEL 1024
#define NROWS   18432   /* B * T * H * W = 2*16*24*24 */
#define HWSZ    576
#define TSZ     16
#define NHEADS  16
#define HDIM    64

#if defined(__has_builtin)
#if __has_builtin(__builtin_amdgcn_global_load_async_to_lds_b128) && \
    __has_builtin(__builtin_amdgcn_s_wait_asynccnt)
#define ASYNC_LDS 1
#endif
#endif
#ifndef ASYNC_LDS
#define ASYNC_LDS 0
#endif

typedef __attribute__((address_space(1))) v4i* gvec4p;
typedef __attribute__((address_space(3))) v4i* lvec4p;

// Copy 16 bytes global -> LDS. Async (ASYNCcnt-tracked, no VGPR round trip)
// when the gfx1250 builtin is available, else plain load+store.
__device__ __forceinline__ void async_copy16(const _Float16* g, _Float16* l) {
#if ASYNC_LDS
  __builtin_amdgcn_global_load_async_to_lds_b128(
      (gvec4p)(void*)g, (lvec4p)(void*)l, 0, 0);
#else
  *(uint4*)l = *(const uint4*)g;
#endif
}
__device__ __forceinline__ void async_wait() {
#if ASYNC_LDS
  __builtin_amdgcn_s_wait_asynccnt(0);
#endif
}

// ---------------------------------------------------------------------------
// WMMA helpers (CDNA5 v_wmma_f32_16x16x32_f16)
// ---------------------------------------------------------------------------
__device__ __forceinline__ v8f wmma_f16(v16h a, v16h b, v8f c) {
  return __builtin_amdgcn_wmma_f32_16x16x32_f16(
      /*neg_a=*/false, a, /*neg_b=*/false, b,
      /*c_mod=*/(short)0, c, /*reuse_a=*/false, /*reuse_b=*/false);
}

__device__ __forceinline__ v8f zero8() {
  v8f z = {0.f,0.f,0.f,0.f,0.f,0.f,0.f,0.f};
  return z;
}

// A fragment: 16x32 f16, row-major source with leading dim `ld` (f16 units).
__device__ __forceinline__ v16h ldsA_frag(const _Float16* base, int ld) {
  int lane = threadIdx.x & 31;
  int m = lane & 15, half = lane >> 4;
  const _Float16* p = base + m * ld + half * 16;
  union { v16h v; uint4 q[2]; } u;
  u.q[0] = *(const uint4*)(p);
  u.q[1] = *(const uint4*)(p + 8);
  return u.v;
}

// B fragment: 32x16 f16 (KxN); source laid out [col][k] with leading dim `ld`.
__device__ __forceinline__ v16h ldsB_frag(const _Float16* baseT, int ld) {
  int lane = threadIdx.x & 31;
  int n = lane & 15, koff = (lane >> 4) * 16;
  const _Float16* p = baseT + n * ld + koff;
  union { v16h v; uint4 q[2]; } u;
  u.q[0] = *(const uint4*)(p);
  u.q[1] = *(const uint4*)(p + 8);
  return u.v;
}

__device__ __forceinline__ float half_max(float v) {
  #pragma unroll
  for (int o = 1; o < 16; o <<= 1) v = fmaxf(v, __shfl_xor(v, o, 32));
  return v;
}
__device__ __forceinline__ float half_sum(float v) {
  #pragma unroll
  for (int o = 1; o < 16; o <<= 1) v += __shfl_xor(v, o, 32);
  return v;
}

// ---------------------------------------------------------------------------
// Elementwise f32 -> f16 (8 elems / thread)
// ---------------------------------------------------------------------------
__global__ void k_f32_to_f16(const float* __restrict__ in,
                             _Float16* __restrict__ out, int n8) {
  int i = blockIdx.x * blockDim.x + threadIdx.x;
  if (i >= n8) return;
  float4 a = ((const float4*)in)[(size_t)i * 2];
  float4 b = ((const float4*)in)[(size_t)i * 2 + 1];
  union { uint4 q; _Float16 e[8]; } u;
  u.e[0] = (_Float16)a.x; u.e[1] = (_Float16)a.y;
  u.e[2] = (_Float16)a.z; u.e[3] = (_Float16)a.w;
  u.e[4] = (_Float16)b.x; u.e[5] = (_Float16)b.y;
  u.e[6] = (_Float16)b.z; u.e[7] = (_Float16)b.w;
  ((uint4*)out)[i] = u.q;
}

// Transpose + convert: out[n*K + k] = (f16) in[k*N + n]
__global__ void k_transpose_f16(const float* __restrict__ in,
                                _Float16* __restrict__ out, int K, int N) {
  int idx = blockIdx.x * blockDim.x + threadIdx.x;
  int K8 = K >> 3;
  if (idx >= N * K8) return;
  int n = idx / K8;
  int k8 = (idx - n * K8) * 8;
  union { uint4 q; _Float16 e[8]; } u;
  #pragma unroll
  for (int i = 0; i < 8; ++i) u.e[i] = (_Float16)in[(size_t)(k8 + i) * N + n];
  *(uint4*)&out[(size_t)n * K + k8] = u.q;
}

// ---------------------------------------------------------------------------
// GEMM: C = A[MxK] * Bt[NxK]^T + bias (+ res), compile-time epilogue flags.
// 128x128x32 tiles, double-buffered LDS with async global->LDS copies,
// 256 threads = 8 waves (4 along M x 2 along N), wave tile 32x64 (2x4 WMMA).
// ---------------------------------------------------------------------------
template <bool RES, bool OUTF, bool OUTH>
__global__ __launch_bounds__(256) void k_gemm_f16(
    const _Float16* __restrict__ A, const _Float16* __restrict__ Bt,
    const float* __restrict__ bias, const float* __restrict__ res,
    float* __restrict__ outF, _Float16* __restrict__ outH,
    int M, int N, int K) {
  __shared__ _Float16 sA[2][128 * 40];
  __shared__ _Float16 sB[2][128 * 40];

  int tid = threadIdx.x;
  int lane = tid & 31, w = tid >> 5;
  int wm = w & 3, wn = w >> 2;               // 4 waves along M, 2 along N
  int bm0 = blockIdx.y * 128, bn0 = blockIdx.x * 128;

  v8f acc[2][4];
  #pragma unroll
  for (int mi = 0; mi < 2; ++mi)
    #pragma unroll
    for (int ni = 0; ni < 4; ++ni) acc[mi][ni] = zero8();

  auto issue = [&](int buf, int kb) {
    #pragma unroll
    for (int p = 0; p < 2; ++p) {
      int idx = tid + p * 256;               // 512 chunks of 8 f16
      int r = idx >> 2, c8 = (idx & 3) * 8;
      async_copy16(&A[(size_t)(bm0 + r) * K + kb + c8], &sA[buf][r * 40 + c8]);
      async_copy16(&Bt[(size_t)(bn0 + r) * K + kb + c8], &sB[buf][r * 40 + c8]);
    }
  };

  issue(0, 0);
  int cur = 0;
  for (int kb = 0; kb < K; kb += 32) {
    async_wait();
    __syncthreads();                          // tile[cur] ready for everyone
    if (kb + 32 < K) issue(cur ^ 1, kb + 32); // prefetch next tile

    v16h af[2], bf[4];
    #pragma unroll
    for (int mi = 0; mi < 2; ++mi)
      af[mi] = ldsA_frag(&sA[cur][(wm * 32 + mi * 16) * 40], 40);
    #pragma unroll
    for (int ni = 0; ni < 4; ++ni)
      bf[ni] = ldsB_frag(&sB[cur][(wn * 64 + ni * 16) * 40], 40);
    #pragma unroll
    for (int mi = 0; mi < 2; ++mi)
      #pragma unroll
      for (int ni = 0; ni < 4; ++ni)
        acc[mi][ni] = wmma_f16(af[mi], bf[ni], acc[mi][ni]);
    cur ^= 1;
  }

  int half = lane >> 4, n = lane & 15;
  #pragma unroll
  for (int mi = 0; mi < 2; ++mi) {
    #pragma unroll
    for (int ni = 0; ni < 4; ++ni) {
      int col = bn0 + wn * 64 + ni * 16 + n;
      float bv = bias[col];
      #pragma unroll
      for (int r = 0; r < 8; ++r) {
        int row = bm0 + wm * 32 + mi * 16 + r + half * 8;
        size_t o = (size_t)row * N + col;
        float v = acc[mi][ni][r] + bv;
        if constexpr (RES)  v += res[o];
        if constexpr (OUTF) outF[o] = v;
        if constexpr (OUTH) outH[o] = (_Float16)v;
      }
    }
  }
}

// ---------------------------------------------------------------------------
// Spatial attention: per (b, t, head), 576 keys, flash-style online softmax.
// Block = 128 threads (4 waves), each wave owns 16 query rows of a 64-row block.
// grid = (9 query blocks, 512 = b*t*h)
// ---------------------------------------------------------------------------
__global__ __launch_bounds__(128) void k_attn_spatial(
    const _Float16* __restrict__ qkv,   // [NROWS x 3072]
    _Float16* __restrict__ outh) {      // [NROWS x 1024]
  __shared__ _Float16 sQ[64 * 72];
  __shared__ _Float16 sK[64 * 72];
  __shared__ _Float16 sVt[64 * 72];     // [d][key]
  __shared__ _Float16 sP[4 * 16 * 72];  // per-wave P scratch

  int tid = threadIdx.x, lane = tid & 31, w = tid >> 5;
  int qb = blockIdx.x;                      // 0..8
  int bth = blockIdx.y;                     // b*256 + t*16 + h
  int h = bth & 15, t = (bth >> 4) & 15, b = bth >> 8;
  int tok0 = b * (TSZ * HWSZ) + t * HWSZ;   // frame base token

  // load 64x64 Q block (async direct-to-LDS)
  #pragma unroll
  for (int p = 0; p < 4; ++p) {
    int idx = tid + p * 128;                // 512 chunks
    int r = idx >> 3, c8 = (idx & 7) * 8;
    size_t g = (size_t)(tok0 + qb * 64 + r) * 3072 + h * 64 + c8;
    async_copy16(&qkv[g], &sQ[r * 72 + c8]);
  }

  v8f O[4];
  #pragma unroll
  for (int di = 0; di < 4; ++di) O[di] = zero8();
  float mrow[8], lrow[8];
  #pragma unroll
  for (int r = 0; r < 8; ++r) { mrow[r] = -1e30f; lrow[r] = 0.f; }
  _Float16* sPw = &sP[w * 16 * 72];
  int half = lane >> 4, n = lane & 15;

  for (int kb = 0; kb < 9; ++kb) {
    // load 64x64 K block (async) and V block transposed (manual scatter)
    #pragma unroll
    for (int p = 0; p < 4; ++p) {
      int idx = tid + p * 128;
      int r = idx >> 3, c8 = (idx & 7) * 8;
      size_t gk = (size_t)(tok0 + kb * 64 + r) * 3072 + 1024 + h * 64 + c8;
      async_copy16(&qkv[gk], &sK[r * 72 + c8]);
      size_t gv = (size_t)(tok0 + kb * 64 + r) * 3072 + 2048 + h * 64 + c8;
      union { uint4 q; _Float16 e[8]; } uv;
      uv.q = *(const uint4*)&qkv[gv];
      #pragma unroll
      for (int i = 0; i < 8; ++i) sVt[(c8 + i) * 72 + r] = uv.e[i];
    }
    async_wait();
    __syncthreads();

    // S = Q * K^T  (16 queries x 64 keys per wave)
    v16h aq0 = ldsA_frag(&sQ[w * 16 * 72], 72);
    v16h aq1 = ldsA_frag(&sQ[w * 16 * 72 + 32], 72);
    v8f S[4];
    #pragma unroll
    for (int ni = 0; ni < 4; ++ni) {
      v16h b0 = ldsB_frag(&sK[ni * 16 * 72], 72);
      v16h b1 = ldsB_frag(&sK[ni * 16 * 72 + 32], 72);
      v8f s = zero8();
      s = wmma_f16(aq0, b0, s);
      s = wmma_f16(aq1, b1, s);
      S[ni] = s;
    }

    // online softmax per row, write P (f16) to per-wave LDS scratch
    #pragma unroll
    for (int r = 0; r < 8; ++r) {
      float s0 = S[0][r] * 0.125f, s1 = S[1][r] * 0.125f;
      float s2 = S[2][r] * 0.125f, s3 = S[3][r] * 0.125f;
      float mx = half_max(fmaxf(fmaxf(s0, s1), fmaxf(s2, s3)));
      float mnew = fmaxf(mrow[r], mx);
      float corr = __expf(mrow[r] - mnew);
      float p0 = __expf(s0 - mnew), p1 = __expf(s1 - mnew);
      float p2 = __expf(s2 - mnew), p3 = __expf(s3 - mnew);
      float ls = half_sum(p0 + p1 + p2 + p3);
      lrow[r] = lrow[r] * corr + ls;
      mrow[r] = mnew;
      #pragma unroll
      for (int di = 0; di < 4; ++di) O[di][r] *= corr;
      int prow = r + half * 8;
      sPw[prow * 72 +  0 + n] = (_Float16)p0;
      sPw[prow * 72 + 16 + n] = (_Float16)p1;
      sPw[prow * 72 + 32 + n] = (_Float16)p2;
      sPw[prow * 72 + 48 + n] = (_Float16)p3;
    }

    // O += P * V
    v16h ap0 = ldsA_frag(sPw, 72);
    v16h ap1 = ldsA_frag(sPw + 32, 72);
    #pragma unroll
    for (int di = 0; di < 4; ++di) {
      v16h b0 = ldsB_frag(&sVt[di * 16 * 72], 72);
      v16h b1 = ldsB_frag(&sVt[di * 16 * 72 + 32], 72);
      O[di] = wmma_f16(ap0, b0, O[di]);
      O[di] = wmma_f16(ap1, b1, O[di]);
    }
    __syncthreads();
  }

  #pragma unroll
  for (int r = 0; r < 8; ++r) {
    float inv = 1.0f / lrow[r];
    int row = qb * 64 + w * 16 + r + half * 8;
    size_t g = (size_t)(tok0 + row) * 1024 + h * 64;
    #pragma unroll
    for (int di = 0; di < 4; ++di)
      outh[g + di * 16 + n] = (_Float16)(O[di][r] * inv);
  }
}

// ---------------------------------------------------------------------------
// Temporal attention: one wave per (b, hw, head); 16x16 causal tile.
// ---------------------------------------------------------------------------
__global__ __launch_bounds__(128) void k_attn_temporal(
    const _Float16* __restrict__ qkv, _Float16* __restrict__ outh) {
  __shared__ _Float16 sQ[4][16 * 72];
  __shared__ _Float16 sK[4][16 * 72];
  __shared__ _Float16 sVt[4][64 * 40];  // [d][key], keys 16..31 zero-padded
  __shared__ _Float16 sP[4][16 * 40];

  int tid = threadIdx.x, lane = tid & 31, w = tid >> 5;
  int gw = blockIdx.x * 4 + w;
  int h = gw & 15;
  int rem = gw >> 4;
  int hw = rem % HWSZ;
  int b = rem / HWSZ;

  _Float16* q_  = sQ[w];
  _Float16* k_  = sK[w];
  _Float16* vt_ = sVt[w];
  _Float16* p_  = sP[w];

  // zero V^T scratch (zero-pads phantom keys 16..31)
  for (int i = lane; i < (64 * 40) / 2; i += 32) ((unsigned*)vt_)[i] = 0u;

  // load Q, K (16x64, async) and V transposed (manual)
  #pragma unroll
  for (int p = 0; p < 4; ++p) {
    int idx = lane + p * 32;           // 128 chunks: 16 rows x 8 chunks
    int t = idx >> 3, c8 = (idx & 7) * 8;
    size_t g = (size_t)(b * (TSZ * HWSZ) + t * HWSZ + hw) * 3072 + h * 64 + c8;
    async_copy16(&qkv[g], &q_[t * 72 + c8]);
    async_copy16(&qkv[g + 1024], &k_[t * 72 + c8]);
    union { uint4 q; _Float16 e[8]; } uv;
    uv.q = *(const uint4*)&qkv[g + 2048];
    #pragma unroll
    for (int i = 0; i < 8; ++i) vt_[(c8 + i) * 40 + t] = uv.e[i];
  }
  async_wait();   // per-wave: Q/K tiles resident before fragment reads

  // S = Q K^T (single 16x16 tile)
  v16h a0 = ldsA_frag(q_, 72), a1 = ldsA_frag(q_ + 32, 72);
  v16h b0 = ldsB_frag(k_, 72), b1 = ldsB_frag(k_ + 32, 72);
  v8f S = zero8();
  S = wmma_f16(a0, b0, S);
  S = wmma_f16(a1, b1, S);

  int half = lane >> 4, n = lane & 15;
  float lsum[8];
  #pragma unroll
  for (int r = 0; r < 8; ++r) {
    int row = r + half * 8;
    float s = S[r] * 0.125f;
    s = (n > row) ? -1e30f : s;        // causal mask
    float mx = half_max(s);
    float e = __expf(s - mx);
    lsum[r] = half_sum(e);
    p_[row * 40 + n] = (_Float16)e;
    p_[row * 40 + 16 + n] = (_Float16)0.f;   // zero phantom keys
  }

  // O = P V  (single K=32 WMMA per 16-wide d tile; keys 16..31 are zero)
  v16h ap = ldsA_frag(p_, 40);
  v8f O[4];
  #pragma unroll
  for (int di = 0; di < 4; ++di) {
    v16h bv = ldsB_frag(&vt_[di * 16 * 40], 40);
    O[di] = wmma_f16(ap, bv, zero8());
  }

  #pragma unroll
  for (int r = 0; r < 8; ++r) {
    float inv = 1.0f / lsum[r];
    int t = r + half * 8;
    size_t g = (size_t)(b * (TSZ * HWSZ) + t * HWSZ + hw) * 1024 + h * 64;
    #pragma unroll
    for (int di = 0; di < 4; ++di)
      outh[g + di * 16 + n] = (_Float16)(O[di][r] * inv);
  }
}

// ---------------------------------------------------------------------------
// Host-side launch
// ---------------------------------------------------------------------------
extern "C" void kernel_launch(void* const* d_in, const int* in_sizes, int n_in,
                              void* d_out, int out_size, void* d_ws, size_t ws_size,
                              hipStream_t stream) {
  const float* x       = (const float*)d_in[0];
  const float* ws_qkv  = (const float*)d_in[1];
  const float* bs_qkv  = (const float*)d_in[2];
  const float* ws_proj = (const float*)d_in[3];
  const float* bs_proj = (const float*)d_in[4];
  const float* wt_qkv  = (const float*)d_in[5];
  const float* bt_qkv  = (const float*)d_in[6];
  const float* wt_proj = (const float*)d_in[7];
  const float* bt_proj = (const float*)d_in[8];
  float* out = (float*)d_out;

  char* ws = (char*)d_ws;
  _Float16* xh      = (_Float16*)ws; ws += (size_t)NROWS * 1024 * 2;
  _Float16* qkvh    = (_Float16*)ws; ws += (size_t)NROWS * 3072 * 2;
  _Float16* attnh   = (_Float16*)ws; ws += (size_t)NROWS * 1024 * 2;
  _Float16* wsqkvT  = (_Float16*)ws; ws += (size_t)3072 * 1024 * 2;
  _Float16* wsprojT = (_Float16*)ws; ws += (size_t)1024 * 1024 * 2;
  _Float16* wtqkvT  = (_Float16*)ws; ws += (size_t)3072 * 1024 * 2;
  _Float16* wtprojT = (_Float16*)ws; ws += (size_t)1024 * 1024 * 2;

  {
    int n8 = NROWS * 1024 / 8;
    k_f32_to_f16<<<(n8 + 255) / 256, 256, 0, stream>>>(x, xh, n8);
  }
  k_transpose_f16<<<(3072 * 128 + 255) / 256, 256, 0, stream>>>(ws_qkv, wsqkvT, 1024, 3072);
  k_transpose_f16<<<(1024 * 128 + 255) / 256, 256, 0, stream>>>(ws_proj, wsprojT, 1024, 1024);
  k_transpose_f16<<<(3072 * 128 + 255) / 256, 256, 0, stream>>>(wt_qkv, wtqkvT, 1024, 3072);
  k_transpose_f16<<<(1024 * 128 + 255) / 256, 256, 0, stream>>>(wt_proj, wtprojT, 1024, 1024);

  dim3 gq(3072 / 128, NROWS / 128);
  dim3 gp(1024 / 128, NROWS / 128);

  // spatial: qkv = x @ Wqkv + b
  k_gemm_f16<false, false, true><<<gq, 256, 0, stream>>>(
      xh, wsqkvT, bs_qkv, nullptr, nullptr, qkvh, NROWS, 3072, 1024);
  k_attn_spatial<<<dim3(9, 512), 128, 0, stream>>>(qkvh, attnh);
  // x1 = x + attn @ Wproj + b  (f32 -> out, f16 -> xh)
  k_gemm_f16<true, true, true><<<gp, 256, 0, stream>>>(
      attnh, wsprojT, bs_proj, x, out, xh, NROWS, 1024, 1024);
  // temporal: qkv = x1 @ Wqkv_t + b
  k_gemm_f16<false, false, true><<<gq, 256, 0, stream>>>(
      xh, wtqkvT, bt_qkv, nullptr, nullptr, qkvh, NROWS, 3072, 1024);
  k_attn_temporal<<<NROWS * NHEADS / 128, 128, 0, stream>>>(qkvh, attnh);
  // out = x1 + attn @ Wproj_t + b
  k_gemm_f16<true, true, false><<<gp, 256, 0, stream>>>(
      attnh, wtprojT, bt_proj, out, out, nullptr, NROWS, 1024, 1024);
}
